// Attention_34634616274985
// MI455X (gfx1250) — matrix operand order repaired and checked by reading, not compiled, not run on hardware
//
#include <hip/hip_runtime.h>

// ---------------------------------------------------------------------------
// CDNA5 (gfx1250) axial attention: bf16 WMMA pipeline, f32 accumulate.
//   K0: x -> bf16            K1/K2: weight transpose+convert
//   K3: QKV GEMM (WMMA, TDM-staged B) with scatter into axial layout
//   K4: per-(b,h,f) 256x256 attention, WMMA QK^T + softmax + WMMA PV
//   K5: output GEMM (WMMA, TDM-staged B) + bias, f32 out
// ---------------------------------------------------------------------------

typedef __attribute__((ext_vector_type(16))) __bf16 v16bf;
typedef __attribute__((ext_vector_type(8)))  __bf16 v8bf;
typedef __attribute__((ext_vector_type(8)))  float  v8f;

#define WMMA_BF16(a, b, c) \
  __builtin_amdgcn_wmma_f32_16x16x32_bf16(false, (a), false, (b), (short)0, (c), false, false)

__device__ __forceinline__ __bf16 f2bf(float f) {
  union { float f; unsigned u; } a; a.f = f;
  unsigned r = (a.u + 0x7FFFu + ((a.u >> 16) & 1u)) >> 16;  // round-nearest-even
  union { unsigned short s; __bf16 b; } o; o.s = (unsigned short)r;
  return o.b;
}

// A fragment 16x32 bf16 (row-major source, ld in elements).
// ISA 7.12.2: lane<16 holds row=lane, K={0..7,16..23}; lane>=16 row=lane-16, K={8..15,24..31}.
__device__ __forceinline__ v16bf load_a_frag(const __bf16* base, int ld, int lane) {
  const int r = lane & 15, h = (lane >> 4) & 1;
  const __bf16* p = base + (size_t)r * ld + h * 8;
  union { v16bf v; v8bf q[2]; } u;
  u.q[0] = *(const v8bf*)(p);
  u.q[1] = *(const v8bf*)(p + 16);
  return u.v;
}

// B fragment 32x16 bf16 gathered from B^T (N-major [N][K], ld in elements).
// ISA: lanes 0-15 hold K={0..7,16..23}, lanes 16-31 hold K={8..15,24..31}, N = lane%16.
__device__ __forceinline__ v16bf load_bt_frag(const __bf16* base, int ld, int lane) {
  const int n = lane & 15, h = (lane >> 4) & 1;
  const __bf16* p = base + (size_t)n * ld + h * 8;
  union { v16bf v; v8bf q[2]; } u;
  u.q[0] = *(const v8bf*)(p);
  u.q[1] = *(const v8bf*)(p + 16);
  return u.v;
}

// ---------------------------------------------------------------------------
// Tensor Data Mover: 2D bf16 tile (tile_d1 rows x tile_d0 elems) global -> LDS.
// D# layout per CDNA5 ISA §8.3 (group0) / §8.4 (group1).
#if __has_builtin(__builtin_amdgcn_tensor_load_to_lds)
#define HAVE_TDM 1
typedef unsigned int tdm_u32x4 __attribute__((ext_vector_type(4)));
typedef int          tdm_i32x4 __attribute__((ext_vector_type(4)));
typedef int          tdm_i32x8 __attribute__((ext_vector_type(8)));

__device__ __forceinline__ void tdm_load_2d_bf16(
    const void* tile_gaddr, unsigned lds_byte_off,
    unsigned tensor_d0, unsigned tensor_d1,
    unsigned tile_d0, unsigned tile_d1, unsigned stride0_elems) {
  unsigned long long ga = (unsigned long long)(uintptr_t)tile_gaddr;
  tdm_u32x4 g0;
  g0[0] = 1u;                                  // count=1, user mode, no gather
  g0[1] = lds_byte_off;                        // lds_addr (bytes)
  g0[2] = (unsigned)ga;                        // global_addr[31:0]
  g0[3] = (unsigned)((ga >> 32) & 0x01FFFFFFull) | 0x80000000u;  // [56:32] | type=2
  tdm_i32x8 g1;
  g1[0] = 0x00010000;                          // data_size=1 (2B); mask/flags=0
  g1[1] = (int)((tensor_d0 & 0xFFFFu) << 16);                 // dim0[15:0] @ 63:48
  g1[2] = (int)((tensor_d0 >> 16) | ((tensor_d1 & 0xFFFFu) << 16)); // dim0[31:16], dim1[15:0]
  g1[3] = (int)((tensor_d1 >> 16) | ((tile_d0 & 0xFFFFu) << 16));   // dim1[31:16], tile_dim0
  g1[4] = (int)(tile_d1 & 0xFFFFu);            // tile_dim1; tile_dim2=0 (2D)
  g1[5] = (int)stride0_elems;                  // tensor_dim0_stride[31:0]
  g1[6] = 0;                                   // stride0[47:32]=0, stride1[15:0]=0
  g1[7] = 0;                                   // stride1[47:16]=0
  tdm_i32x4 z4 = {0, 0, 0, 0};
#if defined(__clang_major__) && (__clang_major__ >= 23)
  tdm_i32x8 z8 = {0, 0, 0, 0, 0, 0, 0, 0};
  __builtin_amdgcn_tensor_load_to_lds(g0, g1, z4, z4, z8, 0);
#else
  __builtin_amdgcn_tensor_load_to_lds(g0, g1, z4, z4, 0);
#endif
}
#endif

// ---------------------------------------------------------------------------
__global__ void cvt_f32_bf16(const float* __restrict__ src, __bf16* __restrict__ dst, int n) {
  int i = blockIdx.x * blockDim.x + threadIdx.x;
  if (i < n) dst[i] = f2bf(src[i]);
}

// src [rows][cols] f32 -> dst [cols][rows] bf16   (builds B^T images for GEMMs)
__global__ void transpose_f32_bf16(const float* __restrict__ src, __bf16* __restrict__ dst,
                                   int rows, int cols) {
  int i = blockIdx.x * blockDim.x + threadIdx.x;
  if (i < rows * cols) {
    int r = i / cols, c = i - r * cols;
    dst[(size_t)c * rows + r] = f2bf(src[i]);
  }
}

// ---------------------------------------------------------------------------
// Shared GEMM core: 8 waves/block, one 128-row group x one 128-col strip.
// B^T slice (128 cols x 64 K) staged in double-buffered LDS: TDM DMA for tile
// i+1 overlaps WMMA on tile i; TENSORcnt is in-order so wait<=1 => tile ready.
#define GEMM_K 512

__device__ __forceinline__ void gemm_core_128x128(
    const __bf16* __restrict__ aBase,   // row-major [*,512], row0 pre-applied
    const __bf16* __restrict__ wT,      // [N][512] B^T image
    int col0, int nRowsT, int lane, int wv,
    __bf16 (*bt)[128][64], v8f acc[8]) {
#if defined(HAVE_TDM)
  // The TDM builtin carries no pointer operand, so the optimizer would see a
  // store-free LDS object and fold its loads to undef.  Install a potential
  // writer: a never-executed store at an opaque index covering all of bt[].
  // This keeps the loads (and their addrspace(3) ds-addressing) intact.
  unsigned never = 0;
  asm volatile("" : "+s"(never));              // opaque zero
  if (never) ((__bf16*)bt)[never & 16383] = f2bf(0.0f);

  if (wv == 0)
    tdm_load_2d_bf16(wT + (size_t)col0 * GEMM_K, 0u, GEMM_K, (unsigned)nRowsT, 64u, 128u, GEMM_K);
  for (int i = 0; i < 8; ++i) {
    const int k0 = i * 64;
    if (wv == 0) {
      if (i < 7) {
        tdm_load_2d_bf16(wT + (size_t)col0 * GEMM_K + (k0 + 64),
                         ((unsigned)(i + 1) & 1u) * 16384u,
                         GEMM_K, (unsigned)nRowsT, 64u, 128u, GEMM_K);
        __builtin_amdgcn_s_wait_tensorcnt(1);   // tile i complete (in-order)
      } else {
        __builtin_amdgcn_s_wait_tensorcnt(0);
      }
    }
    __syncthreads();                            // tile i visible to all waves
    const __bf16 (*B)[64] = bt[i & 1];
#pragma unroll
    for (int kk = 0; kk < 64; kk += 32) {
      v16bf a = load_a_frag(aBase + k0 + kk, GEMM_K, lane);
#pragma unroll
      for (int j = 0; j < 8; ++j) {
        v16bf b = load_bt_frag(&B[j * 16][kk], 64, lane);
        acc[j] = WMMA_BF16(a, b, acc[j]);
      }
    }
    __syncthreads();                            // done reading before next DMA
  }
#else
  for (int i = 0; i < 8; ++i) {
    const int k0 = i * 64;
#pragma unroll
    for (int it = 0; it < 4; ++it) {
      int id = (threadIdx.x & 255) + it * 256;
      int r = id >> 3, cp = id & 7;
      *(v8bf*)&bt[0][r][cp * 8] = *(const v8bf*)&wT[(size_t)(col0 + r) * GEMM_K + k0 + cp * 8];
    }
    __syncthreads();
#pragma unroll
    for (int kk = 0; kk < 64; kk += 32) {
      v16bf a = load_a_frag(aBase + k0 + kk, GEMM_K, lane);
#pragma unroll
      for (int j = 0; j < 8; ++j) {
        v16bf b = load_bt_frag(&bt[0][j * 16][kk], 64, lane);
        acc[j] = WMMA_BF16(a, b, acc[j]);
      }
    }
    __syncthreads();
  }
#endif
}

// ---------------------------------------------------------------------------
// QKV GEMM: C[32768,1536] = xb[32768,512] * W.  Scatter into axial q/k/v:
// q,k = [1024][256][64] (q pre-scaled by 1/8), v = [1024][64][256].
__global__ __launch_bounds__(256) void gemm_qkv_kernel(
    const __bf16* __restrict__ xb, const __bf16* __restrict__ wT,
    __bf16* __restrict__ q_ws, __bf16* __restrict__ k_ws, __bf16* __restrict__ v_ws) {
  __shared__ __bf16 bt[2][128][64];            // 32 KB double buffer
  const int lane = threadIdx.x & 31;
  const int wv   = threadIdx.x >> 5;
  const int mg   = blockIdx.x / 12;
  const int ns   = blockIdx.x % 12;
  const int col0 = ns * 128;
  const int row0 = (mg * 8 + wv) * 16;

  v8f acc[8] = {};
  gemm_core_128x128(xb + (size_t)row0 * GEMM_K, wT, col0, 1536, lane, wv, bt, acc);

  const int n = lane & 15, half = lane >> 4;
#pragma unroll
  for (int j = 0; j < 8; ++j) {
    int e = col0 + j * 16 + n;
    int seg = e >> 9;                  // 0=q, 1=k, 2=v
    int ein = e & 511;
    int h_idx = ein >> 6, d = ein & 63;
#pragma unroll
    for (int r = 0; r < 8; ++r) {
      int g = row0 + r + 8 * half;     // global row: b*8192 + s
      int b_idx = g >> 13;
      int s = g & 8191;
      int f_idx = s >> 8, nsp = s & 255;
      size_t batch = (size_t)((b_idx * 8 + h_idx) * 32 + f_idx);
      float val = acc[j][r];
      if (seg == 0)      q_ws[batch * 16384 + (size_t)nsp * 64 + d] = f2bf(val * 0.125f);
      else if (seg == 1) k_ws[batch * 16384 + (size_t)nsp * 64 + d] = f2bf(val);
      else               v_ws[batch * 16384 + (size_t)d * 256 + nsp] = f2bf(val);
    }
  }
}

// ---------------------------------------------------------------------------
// Attention: one wave = 16 query rows of one (b,h,f) batch. S held in per-wave
// LDS (wave32 DS ops are in-order -> no barriers). dh=64, n_sp=256.
#define ATT_WAVES 4
__global__ __launch_bounds__(32 * ATT_WAVES) void attn_kernel(
    const __bf16* __restrict__ q_ws, const __bf16* __restrict__ k_ws,
    const __bf16* __restrict__ v_ws, __bf16* __restrict__ o_ws) {
  __shared__ float  s_buf[ATT_WAVES][16][256];   // 64 KB
  __shared__ __bf16 p_buf[ATT_WAVES][16][256];   // 32 KB
  const int lane  = threadIdx.x & 31;
  const int wv    = threadIdx.x >> 5;
  const int batch = blockIdx.x >> 2;
  const int rb    = blockIdx.x & 3;
  const int row0  = rb * (16 * ATT_WAVES) + wv * 16;

  const __bf16* Q  = q_ws + (size_t)batch * 16384;
  const __bf16* K  = k_ws + (size_t)batch * 16384;
  const __bf16* Vt = v_ws + (size_t)batch * 16384;   // [64][256] (V transposed)
  float  (*S)[256] = s_buf[wv];
  __bf16 (*P)[256] = p_buf[wv];

  const int n = lane & 15, half = lane >> 4;

  // S = (Q*scale) K^T : Q frags reused across all 16 key tiles
  v16bf aq0 = load_a_frag(Q + (size_t)row0 * 64 + 0, 64, lane);
  v16bf aq1 = load_a_frag(Q + (size_t)row0 * 64 + 32, 64, lane);
#pragma unroll
  for (int nt = 0; nt < 16; ++nt) {
    v8f acc = {};
    v16bf b0 = load_bt_frag(K + (size_t)(nt * 16) * 64 + 0, 64, lane);
    acc = WMMA_BF16(aq0, b0, acc);
    v16bf b1 = load_bt_frag(K + (size_t)(nt * 16) * 64 + 32, 64, lane);
    acc = WMMA_BF16(aq1, b1, acc);
#pragma unroll
    for (int r = 0; r < 8; ++r) S[r + 8 * half][nt * 16 + n] = acc[r];
  }

  // softmax: lane pair (l, l+16) shares row l%16, halves combined via shfl_xor
  const int myrow = lane & 15;
  const int cb = half * 128;
  float m = -3.0e38f;
  for (int c = 0; c < 128; ++c) m = fmaxf(m, S[myrow][cb + c]);
  m = fmaxf(m, __shfl_xor(m, 16, 32));
  float ssum = 0.f;
  for (int c = 0; c < 128; ++c) {
    float e = __expf(S[myrow][cb + c] - m);
    S[myrow][cb + c] = e;
    ssum += e;
  }
  ssum += __shfl_xor(ssum, 16, 32);
  float inv = 1.0f / ssum;
  for (int c = 0; c < 128; ++c) P[myrow][cb + c] = f2bf(S[myrow][cb + c] * inv);

  // O = P V : key dim 256 -> 8 k-steps, dh 64 -> 4 col tiles
  const int f_idx = batch & 31;
  const int h_idx = (batch >> 5) & 7;
  const int b_idx = batch >> 8;
#pragma unroll
  for (int dt = 0; dt < 4; ++dt) {
    v8f acc = {};
#pragma unroll
    for (int kt = 0; kt < 8; ++kt) {
      v16bf a = load_a_frag(&P[0][kt * 32], 256, lane);
      v16bf b = load_bt_frag(Vt + (size_t)(dt * 16) * 256 + kt * 32, 256, lane);
      acc = WMMA_BF16(a, b, acc);
    }
#pragma unroll
    for (int r = 0; r < 8; ++r) {
      int grow = b_idx * 8192 + f_idx * 256 + row0 + r + 8 * half;  // b,(f n)
      o_ws[(size_t)grow * 512 + h_idx * 64 + dt * 16 + n] = f2bf(acc[r]);
    }
  }
}

// ---------------------------------------------------------------------------
// Output GEMM: out[32768,512] = ob[32768,512] * w_out + b_out, f32 result.
__global__ __launch_bounds__(256) void gemm_out_kernel(
    const __bf16* __restrict__ ob, const __bf16* __restrict__ wT,
    const float* __restrict__ bias, float* __restrict__ out) {
  __shared__ __bf16 bt[2][128][64];
  const int lane = threadIdx.x & 31;
  const int wv   = threadIdx.x >> 5;
  const int mg   = blockIdx.x >> 2;
  const int ns   = blockIdx.x & 3;
  const int col0 = ns * 128;
  const int row0 = (mg * 8 + wv) * 16;

  v8f acc[8] = {};
  gemm_core_128x128(ob + (size_t)row0 * GEMM_K, wT, col0, 512, lane, wv, bt, acc);

  const int n = lane & 15, half = lane >> 4;
#pragma unroll
  for (int j = 0; j < 8; ++j) {
    int col = col0 + j * 16 + n;
    float bv = bias[col];
#pragma unroll
    for (int r = 0; r < 8; ++r) {
      int g = row0 + r + 8 * half;
      out[(size_t)g * 512 + col] = acc[j][r] + bv;
    }
  }
}

// ---------------------------------------------------------------------------
extern "C" void kernel_launch(void* const* d_in, const int* in_sizes, int n_in,
                              void* d_out, int out_size, void* d_ws, size_t ws_size,
                              hipStream_t stream) {
  const float* x     = (const float*)d_in[0];   // [4,8192,512]
  const float* w_qkv = (const float*)d_in[1];   // [512,1536]
  const float* w_out = (const float*)d_in[2];   // [512,512]
  const float* b_out = (const float*)d_in[3];   // [512]
  (void)in_sizes; (void)n_in; (void)out_size; (void)ws_size;

  char* ws = (char*)d_ws;
  __bf16* xb   = (__bf16*)(ws);                 //  33,554,432 B  x as bf16
  __bf16* q_ws = (__bf16*)(ws + 33554432ull);   //  33,554,432 B  [1024][256][64]
  __bf16* k_ws = (__bf16*)(ws + 67108864ull);   //  33,554,432 B  [1024][256][64]
  __bf16* v_ws = (__bf16*)(ws + 100663296ull);  //  33,554,432 B  [1024][64][256]
  __bf16* o_ws = (__bf16*)(ws + 134217728ull);  //  33,554,432 B  [32768][512]
  __bf16* wqT  = (__bf16*)(ws + 167772160ull);  //   1,572,864 B  [1536][512]
  __bf16* woT  = (__bf16*)(ws + 169345024ull);  //     524,288 B  [512][512]

  const int NX = 4 * 8192 * 512;  // 16,777,216
  cvt_f32_bf16<<<(NX + 255) / 256, 256, 0, stream>>>(x, xb, NX);
  transpose_f32_bf16<<<(512 * 1536 + 255) / 256, 256, 0, stream>>>(w_qkv, wqT, 512, 1536);
  transpose_f32_bf16<<<(512 * 512 + 255) / 256, 256, 0, stream>>>(w_out, woT, 512, 512);

  gemm_qkv_kernel<<<256 * 12, 256, 0, stream>>>(xb, wqT, q_ws, k_ws, v_ws);
  attn_kernel<<<1024 * 4, 32 * ATT_WAVES, 0, stream>>>(q_ws, k_ws, v_ws, o_ws);
  gemm_out_kernel<<<256 * 4, 256, 0, stream>>>(o_ws, woT, b_out, (float*)d_out);
}
